// RTDETRV2DecoderLayer_43748536877148
// MI455X (gfx1250) — matrix operand order repaired and checked by reading, not compiled, not run on hardware
//
#include <hip/hip_runtime.h>
#include <hip/hip_bf16.h>

typedef __attribute__((ext_vector_type(16))) __bf16 v16bf;
typedef __attribute__((ext_vector_type(8)))  float  v8f;

#define D_MODEL  256
#define N_HEADS  8
#define HEAD_DIM 32
#define N_LEVELS 3
#define N_POINTS 4
#define N_TOTAL  8400
#define BATCH    32
#define LEN_Q    300

// ---------------------------------------------------------------------------
// Pack a 256x256 f32 row-major weight into fragment-ready bf16 order:
//   Wt[((ntile*8 + ktile)*32 + lane)*16 + e] = W[k][n]
//   where k = ktile*32 + (lane>>4)*8 + (e&7) + (e>>3)*16,  n = ntile*16 + (lane&15)
// This matches the CDNA5 16-bit B-matrix (32x16) WMMA lane layout, so each
// lane's B fragment is one contiguous 32-byte chunk (fully coalesced).
// ---------------------------------------------------------------------------
__global__ __launch_bounds__(256)
void pack_w_bf16(const float* __restrict__ W, __bf16* __restrict__ Wt)
{
    const int idx = blockIdx.x * 256 + threadIdx.x;   // 0 .. 65535
    const int e     = idx & 15;
    const int lane  = (idx >> 4) & 31;
    const int ktile = (idx >> 9) & 7;
    const int ntile = idx >> 12;
    const int k = ktile * 32 + ((lane >> 4) << 3) + (e & 7) + ((e >> 3) << 4);
    const int n = ntile * 16 + (lane & 15);
    Wt[idx] = (__bf16)W[k * 256 + n];
}

// ---------------------------------------------------------------------------
// Specialized GEMM: C[M,256] = A[M,256] * W[256,256] + bias  (A,C f32; W is
// pre-packed bf16 fragments). No LDS, no barriers. 256 thr = 8 waves laid out
// 2(M) x 4(N); each wave computes a 32x64 tile: 2 A-frags x 4 B-frags ->
// 8 v_wmma_f32_16x16x32_bf16 per K-step, 8 K-steps. Requires M % 64 == 0.
// A is streamed from HBM exactly once (full-N block).
// ---------------------------------------------------------------------------
__global__ __launch_bounds__(256)
void gemm_n256_wmma(const float* __restrict__ A, const __bf16* __restrict__ Wt,
                    const float* __restrict__ bias, float* __restrict__ C, int M)
{
    const int lane = threadIdx.x & 31;
    const int wave = threadIdx.x >> 5;
    const int wm   = (wave >> 2) * 32;            // 0 / 32 within block
    const int wn   = (wave & 3) * 64;             // 0 / 64 / 128 / 192
    const int bm   = blockIdx.x * 64 + wm;        // this wave's 32-row base

    v8f acc[2][4];
    #pragma unroll
    for (int i = 0; i < 2; ++i)
        #pragma unroll
        for (int j = 0; j < 4; ++j)
            acc[i][j] = (v8f){0.f,0.f,0.f,0.f,0.f,0.f,0.f,0.f};

    const int mrow   = lane & 15;
    const int khalfA = (lane >> 4) << 3;          // 0 / 8 (A 16-bit lane layout)

    const float* a0 = A + (size_t)(bm + mrow) * D_MODEL + khalfA;
    const float* a1 = a0 + (size_t)16 * D_MODEL;
    const int    nt0 = wn >> 4;                   // first of 4 global n-tiles

    for (int kt = 0; kt < 8; ++kt) {
        const int k0 = kt * 32;

        // ---- B fragments: one contiguous 32B load per tile (pre-packed) ----
        v16bf bfrag[4];
        #pragma unroll
        for (int t = 0; t < 4; ++t) {
            const __bf16* p = Wt + ((size_t)((nt0 + t) * 8 + kt) * 32 + lane) * 16;
            bfrag[t] = *reinterpret_cast<const v16bf*>(p);
        }

        // ---- A fragments: 2 m-tiles, each 2x 32B contiguous f32 -> bf16 ----
        v16bf afrag[2];
        #pragma unroll
        for (int mt = 0; mt < 2; ++mt) {
            const float* ap = (mt == 0) ? a0 : a1;
            const float4 x0 = *reinterpret_cast<const float4*>(ap + k0);
            const float4 x1 = *reinterpret_cast<const float4*>(ap + k0 + 4);
            const float4 y0 = *reinterpret_cast<const float4*>(ap + k0 + 16);
            const float4 y1 = *reinterpret_cast<const float4*>(ap + k0 + 20);
            v16bf a;
            a[0]=(__bf16)x0.x; a[1]=(__bf16)x0.y; a[2]=(__bf16)x0.z; a[3]=(__bf16)x0.w;
            a[4]=(__bf16)x1.x; a[5]=(__bf16)x1.y; a[6]=(__bf16)x1.z; a[7]=(__bf16)x1.w;
            a[8]=(__bf16)y0.x; a[9]=(__bf16)y0.y; a[10]=(__bf16)y0.z; a[11]=(__bf16)y0.w;
            a[12]=(__bf16)y1.x; a[13]=(__bf16)y1.y; a[14]=(__bf16)y1.z; a[15]=(__bf16)y1.w;
            afrag[mt] = a;
        }

        #pragma unroll
        for (int mt = 0; mt < 2; ++mt)
            #pragma unroll
            for (int t = 0; t < 4; ++t)
                acc[mt][t] = __builtin_amdgcn_wmma_f32_16x16x32_bf16(
                    false, afrag[mt], false, bfrag[t], (short)0, acc[mt][t],
                    false, false);
    }

    // Epilogue: C/D layout — VGPR r holds M = r + (lane>=16 ? 8:0), N = lane&15.
    const int rbase = bm + ((lane >> 4) << 3);
    #pragma unroll
    for (int t = 0; t < 4; ++t) {
        const int   col = wn + t * 16 + mrow;
        const float bv  = bias[col];
        #pragma unroll
        for (int mt = 0; mt < 2; ++mt) {
            #pragma unroll
            for (int r = 0; r < 8; ++r) {
                C[(size_t)(rbase + mt * 16 + r) * D_MODEL + col] = acc[mt][t][r] + bv;
            }
        }
    }
}

// ---------------------------------------------------------------------------
// Generic GEMM for the small query projections (N = 192 / 96).
// Block tile 128x32, K-step 32, bf16 WMMA. M%128==0, N%32==0, K%32==0.
// ---------------------------------------------------------------------------
__global__ __launch_bounds__(256)
void gemm_bf16_wmma(const float* __restrict__ A, const float* __restrict__ W,
                    const float* __restrict__ bias, float* __restrict__ C,
                    int M, int N, int K)
{
    __shared__ float ldsA[128][33];
    __shared__ float ldsB[32][33];

    const int tid  = threadIdx.x;
    const int lane = tid & 31;
    const int wave = tid >> 5;
    const int bm   = blockIdx.x * 128;
    const int bn   = blockIdx.y * 32;

    v8f c0 = {0.f,0.f,0.f,0.f,0.f,0.f,0.f,0.f};
    v8f c1 = {0.f,0.f,0.f,0.f,0.f,0.f,0.f,0.f};

    const int row    = lane & 15;
    const int khalfA = (lane >> 4) << 4;
    const int khalfB = (lane >> 4) << 4;

    for (int k0 = 0; k0 < K; k0 += 32) {
        #pragma unroll
        for (int i = 0; i < 4; ++i) {
            int s  = tid + i * 256;
            int r  = s >> 3;
            int c4 = (s & 7) << 2;
            const float4 v =
                *reinterpret_cast<const float4*>(&A[(size_t)(bm + r) * K + k0 + c4]);
            ldsA[r][c4 + 0] = v.x; ldsA[r][c4 + 1] = v.y;
            ldsA[r][c4 + 2] = v.z; ldsA[r][c4 + 3] = v.w;
        }
        {
            int r  = tid >> 3;
            int c4 = (tid & 7) << 2;
            const float4 v =
                *reinterpret_cast<const float4*>(&W[(size_t)(k0 + r) * N + bn + c4]);
            ldsB[r][c4 + 0] = v.x; ldsB[r][c4 + 1] = v.y;
            ldsB[r][c4 + 2] = v.z; ldsB[r][c4 + 3] = v.w;
        }
        __syncthreads();

        v16bf afrag, bfrag0, bfrag1;
        #pragma unroll
        for (int e = 0; e < 8; ++e) {
            afrag[e]     = (__bf16)ldsA[wave * 16 + row][khalfA + e];
            afrag[e + 8] = (__bf16)ldsA[wave * 16 + row][khalfA + 8 + e];
        }
        #pragma unroll
        for (int e = 0; e < 16; ++e) {
            bfrag0[e] = (__bf16)ldsB[khalfB + e][row];
            bfrag1[e] = (__bf16)ldsB[khalfB + e][16 + row];
        }

        c0 = __builtin_amdgcn_wmma_f32_16x16x32_bf16(false, afrag, false, bfrag0,
                                                     (short)0, c0, false, false);
        c1 = __builtin_amdgcn_wmma_f32_16x16x32_bf16(false, afrag, false, bfrag1,
                                                     (short)0, c1, false, false);
        __syncthreads();
    }

    const int orow0 = bm + wave * 16 + ((lane >> 4) << 3);
    const int ocol  = bn + row;
    #pragma unroll
    for (int r = 0; r < 8; ++r) {
        const size_t o = (size_t)(orow0 + r) * N + ocol;
        C[o]      = c0[r] + bias[ocol];
        C[o + 16] = c1[r] + bias[ocol + 16];
    }
}

// ---------------------------------------------------------------------------
// Softmax over 12 contiguous elements per (b,q,h) row, in place.
// ---------------------------------------------------------------------------
__global__ __launch_bounds__(256)
void softmax12(float* __restrict__ aw, int nrows)
{
    int i = blockIdx.x * blockDim.x + threadIdx.x;
    if (i >= nrows) return;
    float* p = aw + (size_t)i * 12;
    float m = p[0];
    #pragma unroll
    for (int j = 1; j < 12; ++j) m = fmaxf(m, p[j]);
    float s = 0.f;
    float e[12];
    #pragma unroll
    for (int j = 0; j < 12; ++j) { e[j] = __expf(p[j] - m); s += e[j]; }
    float inv = 1.f / s;
    #pragma unroll
    for (int j = 0; j < 12; ++j) p[j] = e[j] * inv;
}

// ---------------------------------------------------------------------------
// Deformable-attention sampling core: one wave per (b,q,h), lane = channel.
// Each corner gather is one coalesced 128-byte wave access.
// ---------------------------------------------------------------------------
__global__ __launch_bounds__(256)
void ms_deform_sample(const float* __restrict__ value,   // (B, N_TOTAL, 256)
                      const float* __restrict__ refpts,  // (B, Lq, 3, 2)
                      const float* __restrict__ off,     // (B, Lq, 192)
                      const float* __restrict__ aw,      // (B, Lq, 96) softmaxed
                      float* __restrict__ out)           // (B, Lq, 256)
{
    const int gwave = (blockIdx.x * blockDim.x + threadIdx.x) >> 5;
    const int lane  = threadIdx.x & 31;
    if (gwave >= BATCH * LEN_Q * N_HEADS) return;

    const int h  = gwave % N_HEADS;
    const int q  = (gwave / N_HEADS) % LEN_Q;
    const int b  = gwave / (N_HEADS * LEN_Q);
    const int bq = b * LEN_Q + q;

    const int lvl_base[N_LEVELS] = {0, 6400, 8000};
    const int lvl_H[N_LEVELS]    = {80, 40, 20};
    const int lvl_W[N_LEVELS]    = {80, 40, 20};

    const float* vbase = value + (size_t)b * N_TOTAL * D_MODEL + h * HEAD_DIM + lane;
    const float* offp  = off + (size_t)bq * (N_HEADS * N_LEVELS * N_POINTS * 2);
    const float* awp   = aw  + (size_t)bq * (N_HEADS * N_LEVELS * N_POINTS);
    const float* rp    = refpts + (size_t)bq * (N_LEVELS * 2);

    float acc = 0.f;
    #pragma unroll
    for (int l = 0; l < N_LEVELS; ++l) {
        const int   Hh = lvl_H[l], Ww = lvl_W[l];
        const float rx = rp[l * 2 + 0];
        const float ry = rp[l * 2 + 1];
        const float invW = 1.f / (float)Ww;
        const float invH = 1.f / (float)Hh;
        const int   base = lvl_base[l];
        #pragma unroll
        for (int p = 0; p < N_POINTS; ++p) {
            const int   ocol = ((h * N_LEVELS + l) * N_POINTS + p) * 2;
            const float ox = offp[ocol + 0];
            const float oy = offp[ocol + 1];
            const float w  = awp[h * (N_LEVELS * N_POINTS) + l * N_POINTS + p];

            const float lx = rx + ox * invW;
            const float ly = ry + oy * invH;
            const float ix = lx * (float)Ww - 0.5f;
            const float iy = ly * (float)Hh - 0.5f;

            const int ix0 = (int)floorf(ix);
            const int iy0 = (int)floorf(iy);
            const int ix1 = ix0 + 1, iy1 = iy0 + 1;
            const float fx = ix - (float)ix0;
            const float fy = iy - (float)iy0;

            const float mx0 = (ix0 >= 0 && ix0 < Ww) ? 1.f : 0.f;
            const float mx1 = (ix1 >= 0 && ix1 < Ww) ? 1.f : 0.f;
            const float my0 = (iy0 >= 0 && iy0 < Hh) ? 1.f : 0.f;
            const float my1 = (iy1 >= 0 && iy1 < Hh) ? 1.f : 0.f;

            const int x0 = min(max(ix0, 0), Ww - 1);
            const int x1 = min(max(ix1, 0), Ww - 1);
            const int y0 = min(max(iy0, 0), Hh - 1);
            const int y1 = min(max(iy1, 0), Hh - 1);

            const float v00 = vbase[(size_t)(base + y0 * Ww + x0) * D_MODEL] * (mx0 * my0);
            const float v01 = vbase[(size_t)(base + y0 * Ww + x1) * D_MODEL] * (mx1 * my0);
            const float v10 = vbase[(size_t)(base + y1 * Ww + x0) * D_MODEL] * (mx0 * my1);
            const float v11 = vbase[(size_t)(base + y1 * Ww + x1) * D_MODEL] * (mx1 * my1);

            const float s = (1.f - fx) * (1.f - fy) * v00 + fx * (1.f - fy) * v01
                          + (1.f - fx) * fy        * v10 + fx * fy        * v11;
            acc = fmaf(w, s, acc);
        }
    }
    out[(size_t)bq * D_MODEL + h * HEAD_DIM + lane] = acc;
}

// ---------------------------------------------------------------------------
extern "C" void kernel_launch(void* const* d_in, const int* in_sizes, int n_in,
                              void* d_out, int out_size, void* d_ws, size_t ws_size,
                              hipStream_t stream)
{
    const float* query         = (const float*)d_in[0];   // (B, Lq, 256)
    const float* refpts        = (const float*)d_in[1];   // (B, Lq, 3, 2)
    const float* input_flatten = (const float*)d_in[2];   // (B, 8400, 256)
    const float* W_off  = (const float*)d_in[3];          // (256, 192)
    const float* b_off  = (const float*)d_in[4];          // (192)
    const float* W_attn = (const float*)d_in[5];          // (256, 96)
    const float* b_attn = (const float*)d_in[6];          // (96)
    const float* W_val  = (const float*)d_in[7];          // (256, 256)
    const float* b_val  = (const float*)d_in[8];          // (256)
    const float* W_out  = (const float*)d_in[9];          // (256, 256)
    const float* b_out  = (const float*)d_in[10];         // (256)
    float* out = (float*)d_out;                           // (B, Lq, 256)

    const int Mv = BATCH * N_TOTAL;   // 268800
    const int Mq = BATCH * LEN_Q;     // 9600

    // Workspace layout (f32 unless noted): value | off | aw | attn | Wt_val | Wt_out
    float*  value  = (float*)d_ws;                          // Mv * 256
    float*  off    = value + (size_t)Mv * D_MODEL;          // Mq * 192
    float*  aw     = off   + (size_t)Mq * 192;              // Mq * 96
    float*  attn   = aw    + (size_t)Mq * 96;               // Mq * 256
    __bf16* Wt_val = (__bf16*)(attn + (size_t)Mq * D_MODEL);// 65536 bf16
    __bf16* Wt_out = Wt_val + 65536;                        // 65536 bf16

    // 0) pack weights into fragment-ready bf16 (L2-resident, reused by all blocks)
    pack_w_bf16<<<256, 256, 0, stream>>>(W_val, Wt_val);
    pack_w_bf16<<<256, 256, 0, stream>>>(W_out, Wt_out);

    // 1) value projection: full-N blocks, A streamed from HBM exactly once
    gemm_n256_wmma<<<Mv / 64, 256, 0, stream>>>(input_flatten, Wt_val, b_val, value, Mv);

    // 2) sampling offsets: (Mq,256)x(256,192)
    gemm_bf16_wmma<<<dim3(Mq / 128, 192 / 32), 256, 0, stream>>>(
        query, W_off, b_off, off, Mq, 192, D_MODEL);

    // 3) attention logits: (Mq,256)x(256,96)
    gemm_bf16_wmma<<<dim3(Mq / 128, 96 / 32), 256, 0, stream>>>(
        query, W_attn, b_attn, aw, Mq, 96, D_MODEL);

    // 4) softmax over 12 (levels*points) per (b,q,h)
    {
        const int nrows = Mq * N_HEADS;  // 76800
        softmax12<<<(nrows + 255) / 256, 256, 0, stream>>>(aw, nrows);
    }

    // 5) deformable sampling: 1 wave per (b,q,h)
    {
        const int nwaves = BATCH * LEN_Q * N_HEADS;  // 76800
        ms_deform_sample<<<(nwaves + 7) / 8, 256, 0, stream>>>(
            value, refpts, off, aw, attn);
    }

    // 6) output projection
    gemm_n256_wmma<<<Mq / 64, 256, 0, stream>>>(attn, Wt_out, b_out, out, Mq);
}
